// GenerativeUpBlock_81475529605506
// MI455X (gfx1250) — compile-verified, matrix operand order checked by reading, not run on hardware
//
#include <hip/hip_runtime.h>

typedef __attribute__((ext_vector_type(16))) _Float16 v16h;
typedef __attribute__((ext_vector_type(8)))  float    v8f;
typedef __attribute__((ext_vector_type(4)))  unsigned uint32x4;
typedef __attribute__((ext_vector_type(8)))  unsigned uint32x8;

// ---------------------------------------------------------------------------
// Workspace layout (bytes):
//   [0, 524288)        W16frag : W_up as f16 WMMA B-fragments  (64 ct * 8 kt * 32 lanes * 16 h)
//   [524288, 557056)   Wo1frag : Wo1 as f16 WMMA B-fragments   ( 8 ct * 4 kt * 32 lanes * 16 h)
//   [557056, 565248)   hist    : u32[8*256]
//   [565248, 565312)   prefix  : u64[8]   (radix-select prefix -> final threshold key)
//   [565312, 565344)   kcur    : u32[8]
// ---------------------------------------------------------------------------

__device__ __forceinline__ unsigned long long make_key(float p, unsigned m) {
  unsigned u = __float_as_uint(p);
  u = (u & 0x80000000u) ? ~u : (u | 0x80000000u);          // monotonic ascending map
  return ((unsigned long long)u << 32) | (unsigned)(~m);   // tie-break: smaller index wins
}

// --- convert + swizzle weights into WMMA B-fragment layout (f16) ------------
__global__ void prep_weights(const float* __restrict__ W_up, const float* __restrict__ Wo1,
                             _Float16* __restrict__ W16frag, _Float16* __restrict__ Wo1frag) {
  int tid = blockIdx.x * blockDim.x + threadIdx.x;
  if (tid < 64 * 8 * 32) {               // W_up: [256 K x 1024 cols], col = kk*128+d
    int lane = tid & 31;
    int kt   = (tid >> 5) & 7;           // K tile (32 K per tile)
    int ct   = tid >> 8;                 // col tile (16 cols per tile)
    int col  = ct * 16 + (lane & 15);
    int kk = col >> 7, d = col & 127;
    int kb = (lane >= 16) ? 8 : 0;
    _Float16* dst = W16frag + (size_t)tid * 16;
#pragma unroll
    for (int e = 0; e < 16; ++e) {
      int c = kt * 32 + (e >> 3) * 16 + kb + (e & 7);
      dst[e] = (_Float16)W_up[(size_t)kk * 256 * 128 + (size_t)c * 128 + d];
    }
  } else if (tid < 64 * 8 * 32 + 8 * 4 * 32) {   // Wo1: [128 K x 128 cols]
    int t2 = tid - 64 * 8 * 32;
    int lane = t2 & 31;
    int kt   = (t2 >> 5) & 3;
    int ct   = t2 >> 7;
    int col  = ct * 16 + (lane & 15);
    int kb = (lane >= 16) ? 8 : 0;
    _Float16* dst = Wo1frag + (size_t)t2 * 16;
#pragma unroll
    for (int e = 0; e < 16; ++e) {
      int c = kt * 32 + (e >> 3) * 16 + kb + (e & 7);
      dst[e] = (_Float16)Wo1[(size_t)c * 128 + col];
    }
  }
}

// --- GEMM1: child[N x 1024] = feats[N x 256] @ W + b_up  (into d_out rows) --
// Per K-step: issue ALL B-fragment loads first, then run the 8-WMMA chain so
// outstanding loads overlap matrix ops (partial s_wait_loadcnt instead of 0).
__global__ void gemm1_up(const float* __restrict__ feats, const _Float16* __restrict__ Bfrag,
                         const float* __restrict__ b_up, float* __restrict__ out) {
  const int wave = threadIdx.x >> 5;
  const int lane = threadIdx.x & 31;
  const int row0 = blockIdx.x * 16;      // 16 points per block
  const int col0 = wave * 128;           // 8 waves cover 1024 cols
  const int arow = row0 + (lane & 15);
  const int kb   = (lane >= 16) ? 8 : 0;
  v8f acc[8] = {};
  for (int k0 = 0; k0 < 256; k0 += 32) {
    // B fragments for this K-step: 8 tiles, issued as one load burst
    v16h bfr[8];
#pragma unroll
    for (int t = 0; t < 8; ++t)
      bfr[t] = *(const v16h*)(Bfrag + ((size_t)(((col0 >> 4) + t) * 8 + (k0 >> 5)) * 32 + lane) * 16);
    // A fragment: 16 rows x 32 K slice of feats, converted fp32 -> f16
    const float* ap = feats + (size_t)arow * 256 + k0 + kb;
    v16h a;
#pragma unroll
    for (int e = 0; e < 8; ++e) a[e] = (_Float16)ap[e];
#pragma unroll
    for (int e = 0; e < 8; ++e) a[8 + e] = (_Float16)ap[16 + e];
#pragma unroll
    for (int t = 0; t < 8; ++t)
      acc[t] = __builtin_amdgcn_wmma_f32_16x16x32_f16(false, a, false, bfr[t], (short)0, acc[t], false, false);
  }
  const int rofs = (lane >= 16) ? 8 : 0;
#pragma unroll
  for (int t = 0; t < 8; ++t) {
    int colt = col0 + t * 16 + (lane & 15);
    float bias = b_up[colt & 127];
#pragma unroll
    for (int r = 0; r < 8; ++r)
      out[(size_t)(row0 + rofs + r) * 1024 + colt] = acc[t][r] + bias;
  }
}

// --- GEMM2 fused: pred = relu(child @ Wo1 + bo1) @ Wo2 + bo2 ----------------
// Wo1 B-fragments (32 KB) are staged into LDS once per block by the Tensor
// Data Mover (tensor_load_to_lds), then all 8 waves read them via DS loads.
__global__ void gemm2_pred(const float* __restrict__ child, const _Float16* __restrict__ Bfrag,
                           const float* __restrict__ bo1, const float* __restrict__ Wo2,
                           const float* __restrict__ bo2, float* __restrict__ preds) {
  __shared__ _Float16 lB[16384];         // 8 ct * 4 kt * 32 lanes * 16 halves = 32 KB
  const int wave = threadIdx.x >> 5;
  const int lane = threadIdx.x & 31;

  if (wave == 0) {
    // Tensor DMA descriptor (D#), ISA ch.8: 1-row tile of 4096 x 8-byte elems.
    unsigned long long ga = (unsigned long long)(uintptr_t)Bfrag;
    unsigned ldsa = (unsigned)(uintptr_t)(&lB[0]);     // low 32 bits = LDS offset
    uint32x4 g0;
    g0.x = 1u;                                          // count=1, user descriptor
    g0.y = ldsa;                                        // lds_addr
    g0.z = (unsigned)ga;                                // global_addr[31:0]
    g0.w = (unsigned)((ga >> 32) & 0x1FFFFFFu) | (2u << 30);  // global_addr[56:32] | type=2
    uint32x8 g1 = { 0x00030000u,   // data_size=3 (8B), mask=0, no flags
                    0x10000000u,   // tensor_dim0[15:0]=4096 in bits[63:48]
                    0x00010000u,   // tensor_dim0 hi=0, tensor_dim1 lo=1
                    0x10000000u,   // tensor_dim1 hi=0, tile_dim0=4096
                    1u,            // tile_dim1=1, tile_dim2=0
                    4096u,         // tensor_dim0_stride lo
                    0u, 0u };      // stride hi, tensor_dim1_stride=0
    asm volatile("tensor_load_to_lds %0, %1" :: "s"(g0), "s"(g1) : "memory");
    __builtin_amdgcn_s_wait_tensorcnt(0);
  }
  __syncthreads();

  const int row0 = blockIdx.x * 128 + wave * 16;
  const int arow = row0 + (lane & 15);
  const int kb   = (lane >= 16) ? 8 : 0;
  v16h afr[4];
#pragma unroll
  for (int ki = 0; ki < 4; ++ki) {
    const float* ap = child + (size_t)arow * 128 + ki * 32 + kb;
#pragma unroll
    for (int e = 0; e < 8; ++e) afr[ki][e] = (_Float16)ap[e];
#pragma unroll
    for (int e = 0; e < 8; ++e) afr[ki][8 + e] = (_Float16)ap[16 + e];
  }
  float pacc[8];
#pragma unroll
  for (int r = 0; r < 8; ++r) pacc[r] = 0.0f;
#pragma unroll
  for (int t = 0; t < 8; ++t) {
    v8f d = {};
#pragma unroll
    for (int ki = 0; ki < 4; ++ki) {
      const v16h b = *(const v16h*)(lB + (size_t)((t * 4 + ki) * 32 + lane) * 16);
      d = __builtin_amdgcn_wmma_f32_16x16x32_f16(false, afr[ki], false, b, (short)0, d, false, false);
    }
    int colt = t * 16 + (lane & 15);
    float b1 = bo1[colt];
    float w2 = Wo2[colt];
#pragma unroll
    for (int r = 0; r < 8; ++r) {
      float h = d[r] + b1;
      h = h > 0.0f ? h : 0.0f;
      pacc[r] += h * w2;
    }
  }
  float b2 = bo2[0];
#pragma unroll
  for (int r = 0; r < 8; ++r) {
    float v = pacc[r];
#pragma unroll
    for (int ofs = 1; ofs < 16; ofs <<= 1) v += __shfl_xor(v, ofs, 32);
    if ((lane & 15) == 0) preds[row0 + ((lane >= 16) ? 8 : 0) + r] = v + b2;
  }
}

// --- radix select state init ------------------------------------------------
__global__ void sel_init(unsigned* __restrict__ hist, unsigned long long* __restrict__ prefix,
                         unsigned* __restrict__ kcur, const int* __restrict__ kin) {
  int tid = blockIdx.x * blockDim.x + threadIdx.x;
  if (tid < 2048) hist[tid] = 0u;
  if (tid < 8) {
    prefix[tid] = 0ULL;
    int kv = kin[tid];
    kcur[tid] = kv > 0 ? (unsigned)kv : 0u;
  }
}

// --- one radix pass: per-batch histogram of current byte --------------------
__global__ void hist_pass(const float* __restrict__ preds, const int* __restrict__ batch_ids,
                          const unsigned long long* __restrict__ prefix,
                          unsigned* __restrict__ hist, int pass, int M) {
  __shared__ unsigned sh[2048];
  for (int i = threadIdx.x; i < 2048; i += blockDim.x) sh[i] = 0u;
  __syncthreads();
  int m = blockIdx.x * blockDim.x + threadIdx.x;
  if (m < M) {
    int b = batch_ids[m >> 3];
    unsigned long long key = make_key(preds[m], (unsigned)m);
    int shift = 56 - 8 * pass;
    bool match = (pass == 0) || ((key >> (shift + 8)) == (prefix[b] >> (shift + 8)));
    if (match) {
      unsigned v = (unsigned)(key >> shift) & 0xFFu;
      atomicAdd(&sh[b * 256 + v], 1u);
    }
  }
  __syncthreads();
  for (int i = threadIdx.x; i < 2048; i += blockDim.x) {
    unsigned c = sh[i];
    if (c) atomicAdd(&hist[i], c);
  }
}

// --- pick byte containing the k-th largest; reset hist ----------------------
__global__ void scan_pass(unsigned* __restrict__ hist, unsigned long long* __restrict__ prefix,
                          unsigned* __restrict__ kcur, int pass) {
  if (threadIdx.x < 8) {
    int b = threadIdx.x;
    int shift = 56 - 8 * pass;
    unsigned k = kcur[b];
    unsigned chosen = 0;
    for (int v = 255; v >= 0; --v) {
      unsigned c = hist[b * 256 + v];
      if (k > c && v > 0) { k -= c; }
      else { chosen = (unsigned)v; break; }
    }
    prefix[b] |= (unsigned long long)chosen << shift;
    kcur[b] = k;
  }
  __syncthreads();
  for (int i = threadIdx.x; i < 2048; i += blockDim.x) hist[i] = 0u;
}

// --- apply mask: zero pruned rows in place, emit coords + mask --------------
__global__ void finalize(float* __restrict__ out, const int* __restrict__ coords,
                         const int* __restrict__ batch_ids,
                         const unsigned long long* __restrict__ thresh, int M) {
  int gwave = (int)((blockIdx.x * blockDim.x + threadIdx.x) >> 5);
  int lane  = threadIdx.x & 31;
  if (gwave >= M) return;
  int m = gwave;
  int b = batch_ids[m >> 3];
  const float* preds = out + (size_t)M * 128;
  unsigned long long key = make_key(preds[m], (unsigned)m);
  bool sel = key >= thresh[b];
  float* row = out + (size_t)m * 128;
  if (!sel) {
#pragma unroll
    for (int i = 0; i < 4; ++i) row[lane + 32 * i] = 0.0f;
  }
  if (lane == 0) {
    int n = m >> 3, o = m & 7;
    float* upc = out + (size_t)M * 128 + M + (size_t)m * 3;
    float* msk = out + (size_t)M * 128 + M + (size_t)M * 3;
    if (sel) {
      upc[0] = (float)(2 * coords[n * 3 + 0] + ((o >> 2) & 1));
      upc[1] = (float)(2 * coords[n * 3 + 1] + ((o >> 1) & 1));
      upc[2] = (float)(2 * coords[n * 3 + 2] + (o & 1));
    } else {
      upc[0] = 0.0f; upc[1] = 0.0f; upc[2] = 0.0f;
    }
    msk[m] = sel ? 1.0f : 0.0f;
  }
}

extern "C" void kernel_launch(void* const* d_in, const int* in_sizes, int n_in,
                              void* d_out, int out_size, void* d_ws, size_t ws_size,
                              hipStream_t stream) {
  (void)n_in; (void)out_size; (void)ws_size;
  const float* feats     = (const float*)d_in[0];
  const int*   coords    = (const int*)d_in[1];
  const int*   batch_ids = (const int*)d_in[2];
  const int*   kin       = (const int*)d_in[3];
  const float* W_up      = (const float*)d_in[4];
  const float* b_up      = (const float*)d_in[5];
  const float* Wo1       = (const float*)d_in[6];
  const float* bo1       = (const float*)d_in[7];
  const float* Wo2       = (const float*)d_in[8];
  const float* bo2       = (const float*)d_in[9];
  float* out = (float*)d_out;

  const int N = in_sizes[0] / 256;   // 100000 (divisible by 16)
  const int M = N * 8;               // 800000 (divisible by 128)

  char* wsb = (char*)d_ws;
  _Float16* W16frag = (_Float16*)wsb;
  _Float16* Wo1frag = (_Float16*)(wsb + 524288);
  unsigned* hist    = (unsigned*)(wsb + 557056);
  unsigned long long* prefix = (unsigned long long*)(wsb + 565248);
  unsigned* kcur    = (unsigned*)(wsb + 565312);
  float* preds = out + (size_t)M * 128;

  prep_weights<<<68, 256, 0, stream>>>(W_up, Wo1, W16frag, Wo1frag);
  gemm1_up<<<N / 16, 256, 0, stream>>>(feats, W16frag, b_up, out);
  gemm2_pred<<<M / 128, 256, 0, stream>>>(out, Wo1frag, bo1, Wo2, bo2, preds);
  sel_init<<<8, 256, 0, stream>>>(hist, prefix, kcur, kin);
  for (int pass = 0; pass < 8; ++pass) {
    hist_pass<<<(M + 255) / 256, 256, 0, stream>>>(preds, batch_ids, prefix, hist, pass, M);
    scan_pass<<<1, 256, 0, stream>>>(hist, prefix, kcur, pass);
  }
  finalize<<<M / 8, 256, 0, stream>>>(out, coords, batch_ids, prefix, M);
}
